// GCN_16372415332459
// MI455X (gfx1250) — compile-verified, matrix operand order checked by reading, not run on hardware
//
#include <hip/hip_runtime.h>

// ---------------------------------------------------------------------------
// MI455X (gfx1250) GNN block, bf16-everywhere dataflow:
//   elementwise f32->bf16 packing passes, then every GEMM stage moves
//   128x32 bf16 tiles with TENSOR_LOAD_TO_LDS (TDM, double-buffered LDS)
//   and computes with v_wmma_f32_16x16x32_bf16.
// ---------------------------------------------------------------------------

typedef __attribute__((ext_vector_type(16))) __bf16        v16bf;
typedef __attribute__((ext_vector_type(8)))  float         v8f;
typedef __attribute__((ext_vector_type(4)))  unsigned int  v4u;
typedef __attribute__((ext_vector_type(8)))  int           v8i;
typedef __attribute__((ext_vector_type(4)))  int           v4i;

#define EPS_BN 1e-5f
#define LRELU_ALPHA 0.2f

// LDS tile: 128 rows x 32 bf16, TDM-padded to 72B row stride (36 halves)
#define TROW 36
#define TILE_HALVES (128 * TROW)

enum { EPI_NONE = 0, EPI_BNRELU = 1, EPI_LEAKY = 2 };
enum { ST_F32 = 0, ST_BF16 = 1, ST_BF16_T = 2, ST_BF16_DUAL = 3 };

struct GemmArgs {
  const unsigned short* A; const unsigned short* B;
  float* Cf; unsigned short* Cb; unsigned short* Cb2;
  const float* bias; const float* bng; const float* bnb;
  const float* bnm;  const float* bnv;
  int M, N, K;
  int lda, ldb, ldc, ld2;
  int zmod;                                   // batch split: z/zmod, z%zmod
  long long sA1, sA2, sB1, sB2, sC1, sC2;
};

__device__ __forceinline__ unsigned short f32_to_bf16(float f) {
  unsigned int u = __float_as_uint(f);
  u += 0x7FFFu + ((u >> 16) & 1u);            // round-to-nearest-even
  return (unsigned short)(u >> 16);
}

union FragBF16 {
  v16bf v;
  unsigned int u[8];
};

// LDS byte offset of a __shared__ object (generic -> addrspace(3) -> int)
__device__ __forceinline__ unsigned lds_offset_of(const void* p) {
  return (unsigned)(unsigned long long)
      (__attribute__((address_space(3))) const void*)p;
}

// ---------------------------------------------------------------------------
// TDM: load one 128x32 bf16 tile (row stride ld elements in memory) into LDS
// with 8-byte padding after each 64-byte row (pad_interval=16DW, amount=2DW).
// Issued by one wave; tracked with TENSORcnt.
// ---------------------------------------------------------------------------
__device__ __forceinline__ void tdm_load_tile(unsigned lds_off,
                                              const unsigned short* gptr,
                                              unsigned ld_elems) {
  unsigned long long ga = (unsigned long long)gptr;
  v4u g0;
  g0[0] = 1u;                                          // count=1, user D#
  g0[1] = lds_off;                                     // LDS byte address
  g0[2] = (unsigned)ga;                                // global_addr[31:0]
  g0[3] = (unsigned)((ga >> 32) & 0x01FFFFFFu) | (2u << 30); // addr[56:32]|type=2

  unsigned long long st0 = (unsigned long long)ld_elems;
  v8i g1;
  // data_size=1(2B) | pad_enable | pad_interval=3(16DW) | pad_amount=1(2DW)
  g1[0] = (int)((1u << 16) | (1u << 20) | (3u << 22) | (1u << 25));
  g1[1] = 0;                                           // td0 lo16 in [63:48] = 0
  g1[2] = (int)0x4000u;                                // td0=0x40000000 (no OOB)
  g1[3] = (int)(0x4000u | (32u << 16));                // td1=0x40000000, tile_dim0=32
  g1[4] = 128;                                         // tile_dim1=128, tile_dim2=0
  g1[5] = (int)(st0 & 0xFFFFFFFFu);                    // tensor_dim0_stride lo32
  g1[6] = (int)((st0 >> 32) & 0xFFFFu);                // stride hi16
  g1[7] = 0;
  v4i z4 = {0, 0, 0, 0};
  v8i z8 = {0, 0, 0, 0, 0, 0, 0, 0};
  __builtin_amdgcn_tensor_load_to_lds(g0, g1, z4, z4, z8, 0);
}

// ---------------------------------------------------------------------------
// Tiled WMMA GEMM, all operands bf16 [rows][ld] with B pre-stored as [n][k].
// 128x128 block tile, BK=32, 8 waves (4Mx2N), 2x4 WMMAs per wave per K-step.
// CONV=1: A tile start is tap-shifted into the zero-guard-padded o buffer.
// ---------------------------------------------------------------------------
template<int CONV, int EPI, int ST>
__global__ void __launch_bounds__(256)
wmma_gemm(GemmArgs g) {
  __shared__ __align__(16) unsigned short lsA[2][TILE_HALVES];
  __shared__ __align__(16) unsigned short lsB[2][TILE_HALVES];

  const int tid   = threadIdx.x;
  const int lane  = tid & 31;
  const int wave  = tid >> 5;
  const int waveM = wave & 3;
  const int waveN = wave >> 2;
  const int bm0   = blockIdx.y * 128;
  const int bn0   = blockIdx.x * 128;
  const int z     = blockIdx.z;

  const long long aOff = (long long)(z / g.zmod) * g.sA1 + (long long)(z % g.zmod) * g.sA2;
  const long long bOff = (long long)(z / g.zmod) * g.sB1 + (long long)(z % g.zmod) * g.sB2;
  const long long cOff = (long long)(z / g.zmod) * g.sC1 + (long long)(z % g.zmod) * g.sC2;

  const unsigned short* Abase = g.A + aOff;
  const unsigned short* Bbase = g.B + bOff + (long long)bn0 * g.ldb;

  const unsigned offA0 = lds_offset_of(&lsA[0][0]);
  const unsigned offA1 = lds_offset_of(&lsA[1][0]);
  const unsigned offB0 = lds_offset_of(&lsB[0][0]);
  const unsigned offB1 = lds_offset_of(&lsB[1][0]);

  v8f acc[2][4];
  const v8f vzero = {0.f, 0.f, 0.f, 0.f, 0.f, 0.f, 0.f, 0.f};
#pragma unroll
  for (int i = 0; i < 2; ++i)
#pragma unroll
    for (int j = 0; j < 4; ++j) acc[i][j] = vzero;

  // V_WMMA_F32_16X16X32_BF16 16-bit operand layout:
  // lanes 0-15: row = lane, K = {0..7,16..23}; lanes 16-31: K = {8..15,24..31}
  const int kb   = (lane >> 4) * 8;
  const int lrow = lane & 15;
  const int nk   = g.K / 32;

  // prologue: tile 0 into buffer 0
  if (wave == 0) {
    const unsigned short* pa = Abase + (long long)bm0 * g.lda;  // k0=0: tap0/ci0
    tdm_load_tile(offA0, pa, (unsigned)g.lda);
    tdm_load_tile(offB0, Bbase, (unsigned)g.ldb);
    __builtin_amdgcn_s_wait_tensorcnt(0);
  }
  __syncthreads();

  for (int t = 0; t < nk; ++t) {
    const int buf = t & 1;
    const bool pre = (t + 1) < nk;

    if (wave == 0 && pre) {                   // issue next tiles into buf^1
      const int k0n = (t + 1) * 32;
      const unsigned short* pa;
      if (CONV) {
        int ks = k0n >> 9, ci = k0n & 511;    // tap index, channel offset
        pa = Abase + (long long)(bm0 + ks) * g.lda + ci;
      } else {
        pa = Abase + (long long)bm0 * g.lda + k0n;
      }
      tdm_load_tile(buf ? offA0 : offA1, pa, (unsigned)g.lda);
      tdm_load_tile(buf ? offB0 : offB1, Bbase + k0n, (unsigned)g.ldb);
    }

    const unsigned short* tA = &lsA[buf][0];
    const unsigned short* tB = &lsB[buf][0];

    FragBF16 fa[2], fb[4];
#pragma unroll
    for (int mt = 0; mt < 2; ++mt) {
      const unsigned short* pr = tA + (waveM * 32 + mt * 16 + lrow) * TROW;
#pragma unroll
      for (int i = 0; i < 8; ++i) {
        int k = (i < 4) ? (kb + 2 * i) : (16 + kb + 2 * (i - 4));
        fa[mt].u[i] = *(const unsigned int*)(pr + k);
      }
    }
#pragma unroll
    for (int nt = 0; nt < 4; ++nt) {
      const unsigned short* pr = tB + (waveN * 64 + nt * 16 + lrow) * TROW;
#pragma unroll
      for (int i = 0; i < 8; ++i) {
        int k = (i < 4) ? (kb + 2 * i) : (16 + kb + 2 * (i - 4));
        fb[nt].u[i] = *(const unsigned int*)(pr + k);
      }
    }

#pragma unroll
    for (int mt = 0; mt < 2; ++mt)
#pragma unroll
      for (int nt = 0; nt < 4; ++nt)
        acc[mt][nt] = __builtin_amdgcn_wmma_f32_16x16x32_bf16(
            false, fa[mt].v, false, fb[nt].v, (short)0, acc[mt][nt], false, false);

    if (wave == 0 && pre) __builtin_amdgcn_s_wait_tensorcnt(0);
    __syncthreads();
  }

  // ---- epilogue + store (C/D layout: VGPR r -> M=r(+8 if lane>=16), N=lane&15)
  const int rowHi = (lane >> 4) * 8;
  const int col   = lane & 15;
#pragma unroll
  for (int mt = 0; mt < 2; ++mt) {
#pragma unroll
    for (int nt = 0; nt < 4; ++nt) {
      const int n = bn0 + waveN * 64 + nt * 16 + col;
      float sc = 1.f, off = 0.f;
      if (EPI == EPI_BNRELU) {
        sc  = g.bng[n] * rsqrtf(g.bnv[n] + EPS_BN);
        off = (g.bias[n] - g.bnm[n]) * sc + g.bnb[n];
      }
#pragma unroll
      for (int r = 0; r < 8; ++r) {
        const int m = bm0 + waveM * 32 + mt * 16 + rowHi + r;
        float v = acc[mt][nt][r];
        if (EPI == EPI_BNRELU)     { v = v * sc + off; v = v > 0.f ? v : 0.f; }
        else if (EPI == EPI_LEAKY) { v = v > 0.f ? v : LRELU_ALPHA * v; }
        if (ST == ST_F32) {
          g.Cf[cOff + (long long)m * g.ldc + n] = v;
        } else if (ST == ST_BF16) {
          g.Cb[cOff + (long long)m * g.ldc + n] = f32_to_bf16(v);
        } else if (ST == ST_BF16_T) {          // split m=(b,node), store [b][n][node]
          g.Cb[(long long)(m >> 9) * g.sC1 + (long long)n * g.ldc + (m & 511)] =
              f32_to_bf16(v);
        } else {                               // dual: normal + transposed copy
          unsigned short h = f32_to_bf16(v);
          g.Cb [cOff + (long long)m * g.ldc + n] = h;
          g.Cb2[cOff + (long long)n * g.ld2 + m] = h;
        }
      }
    }
  }
}

// ---------------------------------------------------------------------------
// Row softmax over 512 logits: one wave32 per row, f32 in -> bf16 probs out.
// ---------------------------------------------------------------------------
__global__ void __launch_bounds__(256)
softmax_rows(const float* e, unsigned short* att, long long nRows) {
  const int lane = threadIdx.x & 31;
  const int w    = threadIdx.x >> 5;
  long long row  = (long long)blockIdx.x * 8 + w;
  if (row >= nRows) return;
  const float* p = e + row * 512;
  unsigned short* q = att + row * 512;

  float vals[16];
  float mx = -3.0e38f;
#pragma unroll
  for (int i = 0; i < 16; ++i) { vals[i] = p[lane + i * 32]; mx = fmaxf(mx, vals[i]); }
#pragma unroll
  for (int d = 16; d > 0; d >>= 1) mx = fmaxf(mx, __shfl_xor(mx, d, 32));
  float s = 0.f;
#pragma unroll
  for (int i = 0; i < 16; ++i) { vals[i] = __expf(vals[i] - mx); s += vals[i]; }
#pragma unroll
  for (int d = 16; d > 0; d >>= 1) s += __shfl_xor(s, d, 32);
  float inv = 1.f / s;
#pragma unroll
  for (int i = 0; i < 16; ++i) q[lane + i * 32] = f32_to_bf16(vals[i] * inv);
}

// ---------------------------------------------------------------------------
// Elementwise packing kernels (f32 -> bf16, with layout changes)
// ---------------------------------------------------------------------------
__global__ void cvt_flat(const float* in, unsigned short* out, long long n) {
  long long i = (long long)blockIdx.x * 256 + threadIdx.x;
  if (i < n) out[i] = f32_to_bf16(in[i]);
}

__global__ void cvt_transpose(const float* in, unsigned short* out, int R, int C) {
  long long i = (long long)blockIdx.x * 256 + threadIdx.x;
  if (i < (long long)R * C) {
    int r = (int)(i / C), c = (int)(i % C);
    out[(long long)c * R + r] = f32_to_bf16(in[i]);
  }
}

// conv_w[co][ci][k] (stride: ((co*512)+ci)*3+k) -> convwT[co][k*512+ci]
__global__ void cvt_convw(const float* w, unsigned short* out) {
  long long i = (long long)blockIdx.x * 256 + threadIdx.x;
  if (i < (long long)512 * 1536) {
    int co = (int)(i / 1536), rem = (int)(i % 1536);
    int ks = rem >> 9, ci = rem & 511;
    out[i] = f32_to_bf16(w[((long long)co * 512 + ci) * 3 + ks]);
  }
}

// zero guard rows 0 and 513 of oPad[b][514][512]
__global__ void zero_guard(unsigned short* oPad) {
  long long i = (long long)blockIdx.x * 256 + threadIdx.x;
  if (i < 32LL * 2 * 512) {
    int b = (int)(i >> 10), r = (int)((i >> 9) & 1), c = (int)(i & 511);
    oPad[(long long)b * 514 * 512 + (r ? 513LL * 512 : 0) + c] = 0;
  }
}

// ---------------------------------------------------------------------------
// Host launcher
// ---------------------------------------------------------------------------
extern "C" void kernel_launch(void* const* d_in, const int* in_sizes, int n_in,
                              void* d_out, int out_size, void* d_ws, size_t ws_size,
                              hipStream_t stream) {
  const float* x     = (const float*)d_in[0];
  const float* adj   = (const float*)d_in[1];
  const float* gc_w  = (const float*)d_in[2];
  const float* gc_b  = (const float*)d_in[3];
  const float* bn1_g = (const float*)d_in[4];
  const float* bn1_b = (const float*)d_in[5];
  const float* bn1_m = (const float*)d_in[6];
  const float* bn1_v = (const float*)d_in[7];
  const float* gat_w = (const float*)d_in[8];
  const float* conv_w= (const float*)d_in[9];
  const float* conv_b= (const float*)d_in[10];
  const float* bn2_g = (const float*)d_in[11];
  const float* bn2_b = (const float*)d_in[12];
  const float* bn2_m = (const float*)d_in[13];
  const float* bn2_v = (const float*)d_in[14];
  float* out = (float*)d_out;

  const int B = 32, N = 512, F_IN = 256, HID = 512, OUT = 512, H = 4, D = 128;

  // workspace carve-up (bf16 operands + f32 logits)
  char* ws = (char*)d_ws;
  size_t off = 0;
  auto take = [&](size_t bytes) { char* p = ws + off; off += (bytes + 255) & ~(size_t)255; return p; };
  unsigned short* xb    = (unsigned short*)take((size_t)B * N * F_IN * 2);   // [B*N][256]
  unsigned short* adjb  = (unsigned short*)take((size_t)N * N * 2);          // [512][512]
  unsigned short* gcwT  = (unsigned short*)take((size_t)HID * F_IN * 2);     // [512][256]
  unsigned short* gatb  = (unsigned short*)take((size_t)H * D * HID * 2);    // [h][d][f]
  unsigned short* cwT   = (unsigned short*)take((size_t)OUT * 3 * OUT * 2);  // [co][1536]
  unsigned short* supT  = (unsigned short*)take((size_t)B * HID * N * 2);    // [b][d][node]
  unsigned short* gbuf  = (unsigned short*)take((size_t)B * N * HID * 2);    // [b][node][d]
  unsigned short* hh    = (unsigned short*)take((size_t)B * H * N * D * 2);  // [z][node][d]
  unsigned short* hhT   = (unsigned short*)take((size_t)B * H * D * N * 2);  // [z][d][node]
  float*          e     = (float*)take((size_t)B * H * N * N * 4);           // logits
  unsigned short* attb  = (unsigned short*)take((size_t)B * H * N * N * 2);  // probs
  unsigned short* oPad  = (unsigned short*)take((size_t)B * (N + 2) * OUT * 2);

  // ---- packing pre-passes ----
  cvt_flat<<<dim3((B * N * F_IN + 255) / 256), 256, 0, stream>>>(x, xb, (long long)B * N * F_IN);
  cvt_flat<<<dim3((N * N + 255) / 256), 256, 0, stream>>>(adj, adjb, (long long)N * N);
  cvt_flat<<<dim3((H * D * HID + 255) / 256), 256, 0, stream>>>(gat_w, gatb, (long long)H * D * HID);
  cvt_transpose<<<dim3((F_IN * HID + 255) / 256), 256, 0, stream>>>(gc_w, gcwT, F_IN, HID);
  cvt_convw<<<dim3((512 * 1536 + 255) / 256), 256, 0, stream>>>(conv_w, cwT);
  zero_guard<<<dim3(128), 256, 0, stream>>>(oPad);

  GemmArgs a;

  // ---- Stage 1: support^T[b][d][node] = (x @ gc_w)^T ----
  a = {};
  a.A = xb; a.B = gcwT; a.Cb = supT;
  a.M = B * N; a.N = HID; a.K = F_IN;
  a.lda = F_IN; a.ldb = F_IN; a.ldc = N;     // transposed store: row stride N
  a.zmod = 1; a.sC1 = (long long)HID * N;    // per-b block in supT
  wmma_gemm<0, EPI_NONE, ST_BF16_T>
      <<<dim3(HID / 128, (B * N) / 128, 1), 256, 0, stream>>>(a);

  // ---- Stage 2: g[b] = relu(BN1(adj @ support[b] + gc_b)), bf16 ----
  a = {};
  a.A = adjb; a.B = supT; a.Cb = gbuf;
  a.bias = gc_b; a.bng = bn1_g; a.bnb = bn1_b; a.bnm = bn1_m; a.bnv = bn1_v;
  a.M = N; a.N = HID; a.K = N;
  a.lda = N; a.ldb = N; a.ldc = HID;
  a.zmod = 1; a.sB1 = (long long)HID * N; a.sC1 = (long long)N * HID;
  wmma_gemm<0, EPI_BNRELU, ST_BF16>
      <<<dim3(HID / 128, N / 128, B), 256, 0, stream>>>(a);

  // ---- Stage 3: hh[b,h] = g[b] @ gat_w[h]^T, dual-stored (hh and hh^T) ----
  a = {};
  a.A = gbuf; a.B = gatb; a.Cb = hh; a.Cb2 = hhT;
  a.M = N; a.N = D; a.K = HID;
  a.lda = HID; a.ldb = HID; a.ldc = D; a.ld2 = N;
  a.zmod = H;
  a.sA1 = (long long)N * HID;     a.sA2 = 0;
  a.sB1 = 0;                      a.sB2 = (long long)D * HID;
  a.sC1 = (long long)H * N * D;   a.sC2 = (long long)N * D;
  wmma_gemm<0, EPI_NONE, ST_BF16_DUAL>
      <<<dim3(1, N / 128, B * H), 256, 0, stream>>>(a);

  // ---- Stage 4a: e[z] = leakyrelu(hh[z] @ hh[z]^T), f32 logits ----
  a = {};
  a.A = hh; a.B = hh; a.Cf = e;
  a.M = N; a.N = N; a.K = D;
  a.lda = D; a.ldb = D; a.ldc = N;
  a.zmod = 1;
  a.sA1 = (long long)N * D; a.sB1 = (long long)N * D; a.sC1 = (long long)N * N;
  wmma_gemm<0, EPI_LEAKY, ST_F32>
      <<<dim3(N / 128, N / 128, B * H), 256, 0, stream>>>(a);

  // ---- Stage 4b: softmax rows, bf16 probabilities ----
  {
    long long rows = (long long)B * H * N;
    softmax_rows<<<dim3((unsigned)((rows + 7) / 8)), 256, 0, stream>>>(e, attb, rows);
  }

  // ---- Stage 4c: oPad[b][1+node][h*D:] = att[z] @ hh[z] ----
  a = {};
  a.A = attb; a.B = hhT; a.Cb = oPad + OUT;  // +OUT: skip leading guard row
  a.M = N; a.N = D; a.K = N;
  a.lda = N; a.ldb = N; a.ldc = OUT;
  a.zmod = H;
  a.sA1 = (long long)H * N * N;  a.sA2 = (long long)N * N;
  a.sB1 = (long long)H * D * N;  a.sB2 = (long long)D * N;
  a.sC1 = (long long)(N + 2) * OUT; a.sC2 = (long long)D;
  wmma_gemm<0, EPI_NONE, ST_BF16>
      <<<dim3(1, N / 128, B * H), 256, 0, stream>>>(a);

  // ---- Stage 5: y[b] = relu(BN2(conv1d(o[b]) + conv_b)), tap-shifted tiles ----
  a = {};
  a.A = oPad; a.B = cwT; a.Cf = out;
  a.bias = conv_b; a.bng = bn2_g; a.bnb = bn2_b; a.bnm = bn2_m; a.bnv = bn2_v;
  a.M = N; a.N = OUT; a.K = 3 * OUT;         // K = 1536 (3 taps x 512 ch)
  a.lda = OUT; a.ldb = 3 * OUT; a.ldc = OUT;
  a.zmod = 1;
  a.sA1 = (long long)(N + 2) * OUT; a.sC1 = (long long)N * OUT;
  wmma_gemm<1, EPI_BNRELU, ST_F32>
      <<<dim3(OUT / 128, N / 128, B), 256, 0, stream>>>(a);
}
